// InfoNCE_4217657884717
// MI455X (gfx1250) — compile-verified, matrix operand order checked by reading, not run on hardware
//
#include <hip/hip_runtime.h>
#include <math.h>

#define TEMPERATURE 0.07f
#define EPSN 1e-12f
#define B_DIM 16384
#define D_DIM 512

// ---- fp32 fallback tile params ----
#define KC 32
#define LDA 36   // padded LDS row stride (floats)

// ---- bf16x3 tile params ----
#define KCB 32
#define LDH 40                  // padded LDS row stride (halves): 80B rows, 16B aligned
#define TILE_ELEMS (64 * LDH)   // halves per staged array (64 rows)
#define ARR_BYTES (TILE_ELEMS * 2)
#define BUF_BYTES (4 * ARR_BYTES)   // one double-buffer slot: Ahi|Alo|Bhi|Blo
#define NCHUNK (D_DIM / KCB)

typedef float v2f  __attribute__((ext_vector_type(2)));
typedef float v8f  __attribute__((ext_vector_type(8)));
typedef __bf16 v8bf  __attribute__((ext_vector_type(8)));
typedef __bf16 v16bf __attribute__((ext_vector_type(16)));

// ---------------------------------------------------------------------------
// bf16 helpers (RNE split: x = hi + lo, both bf16; x - (float)hi is exact fp32)
// ---------------------------------------------------------------------------
__device__ __forceinline__ unsigned f2bf_rne(float x) {
  unsigned u = __float_as_uint(x);
  return (u + 0x7FFFu + ((u >> 16) & 1u)) >> 16;
}
__device__ __forceinline__ float bf2f(unsigned h) {
  return __uint_as_float(h << 16);
}
__device__ __forceinline__ void split4(float4 v, uint2& hi, uint2& lo) {
  unsigned h0 = f2bf_rne(v.x), h1 = f2bf_rne(v.y), h2 = f2bf_rne(v.z), h3 = f2bf_rne(v.w);
  unsigned l0 = f2bf_rne(v.x - bf2f(h0));
  unsigned l1 = f2bf_rne(v.y - bf2f(h1));
  unsigned l2 = f2bf_rne(v.z - bf2f(h2));
  unsigned l3 = f2bf_rne(v.w - bf2f(h3));
  hi.x = h0 | (h1 << 16); hi.y = h2 | (h3 << 16);
  lo.x = l0 | (l1 << 16); lo.y = l2 | (l3 << 16);
}
// Fragment load: ISA 16-bit A/B layout = two contiguous 16B pieces per lane
__device__ __forceinline__ v16bf ldfrag(const unsigned short* p) {
  v8bf a = *(const v8bf*)p;
  v8bf b = *(const v8bf*)(p + 16);
  return __builtin_shufflevector(a, b, 0, 1, 2, 3, 4, 5, 6, 7,
                                 8, 9, 10, 11, 12, 13, 14, 15);
}
// CDNA5 async DMA: global -> LDS, 16B per lane, tracked by ASYNCcnt.
// LDS operand = wave-relative byte offset (generic shared addr truncated to 32b).
__device__ __forceinline__ void async_cp16(unsigned ldsByteOff, const unsigned short* g) {
  asm volatile("global_load_async_to_lds_b128 %0, %1, off"
               :: "v"(ldsByteOff), "v"(g) : "memory");
}
__device__ __forceinline__ void wait_async0() {
  asm volatile("s_wait_asynccnt 0" ::: "memory");
}

// ---------------------------------------------------------------------------
// Kernel 1: per-row inverse norms (1/0.07 folded into P's scale) + zero sums.
// ---------------------------------------------------------------------------
__global__ __launch_bounds__(256) void infonce_prep(
    const float* __restrict__ P, const float* __restrict__ Tm,
    float* __restrict__ sA, float* __restrict__ sB,
    float* __restrict__ rowsum, float* __restrict__ colsum) {
  int wave = threadIdx.x >> 5;
  int lane = threadIdx.x & 31;
  int row  = blockIdx.x * 8 + wave;
  const float4* p4 = (const float4*)(P  + (size_t)row * D_DIM);
  const float4* t4 = (const float4*)(Tm + (size_t)row * D_DIM);
  float accp = 0.f, acct = 0.f;
  for (int i = lane; i < D_DIM / 4; i += 32) {
    float4 a = p4[i]; accp += a.x * a.x + a.y * a.y + a.z * a.z + a.w * a.w;
    float4 b = t4[i]; acct += b.x * b.x + b.y * b.y + b.z * b.z + b.w * b.w;
  }
  for (int off = 16; off > 0; off >>= 1) {
    accp += __shfl_xor(accp, off, 32);
    acct += __shfl_xor(acct, off, 32);
  }
  if (lane == 0) {
    float np = sqrtf(accp), nt = sqrtf(acct);
    sA[row] = 1.0f / (fmaxf(np, EPSN) * TEMPERATURE);
    sB[row] = 1.0f / fmaxf(nt, EPSN);
    rowsum[row] = 0.f;
    colsum[row] = 0.f;
  }
}

// ---------------------------------------------------------------------------
// Kernel 2a: one-time bf16 hi/lo split of both matrices (each element once).
// ---------------------------------------------------------------------------
__global__ __launch_bounds__(256) void infonce_split(
    const float* __restrict__ P, const float* __restrict__ Tm,
    unsigned short* __restrict__ Phi, unsigned short* __restrict__ Plo,
    unsigned short* __restrict__ Thi, unsigned short* __restrict__ Tlo) {
  size_t i = (size_t)blockIdx.x * 256 + threadIdx.x;   // float4 index
  uint2 hi, lo;
  split4(((const float4*)P)[i], hi, lo);
  ((uint2*)Phi)[i] = hi;  ((uint2*)Plo)[i] = lo;
  split4(((const float4*)Tm)[i], hi, lo);
  ((uint2*)Thi)[i] = hi;  ((uint2*)Tlo)[i] = lo;
}

// ---------------------------------------------------------------------------
// Kernel 2b: bf16x3 tiled GEMM via V_WMMA_F32_16X16X32_BF16.
// 256 threads = 8 waves, macro tile 64x64; wave owns a 16x32 strip (2 accums).
// Staging: double-buffered GLOBAL_LOAD_ASYNC_TO_LDS_B128 (ASYNCcnt DMA) so the
// chunk-(n+1) copy overlaps chunk-n WMMAs; one barrier per chunk.
// logits = (aH+aL)(bH+bL) ~= aH*bH + aH*bL + aL*bH  (aL*bL ~ 2^-17, dropped)
// ---------------------------------------------------------------------------
__global__ __launch_bounds__(256) void infonce_tile_bf16(
    const unsigned short* __restrict__ Phi, const unsigned short* __restrict__ Plo,
    const unsigned short* __restrict__ Thi, const unsigned short* __restrict__ Tlo,
    const float* __restrict__ sA, const float* __restrict__ sB,
    float* __restrict__ rowsum, float* __restrict__ colsum,
    float* __restrict__ diag) {
  // [buf][arr][64*LDH] halves; arr: 0=Ahi 1=Alo 2=Bhi 3=Blo  (40KB total)
  __shared__ unsigned short smem[2 * 4 * TILE_ELEMS];

  const int tid  = threadIdx.x;
  const int lane = tid & 31;
  const int wave = tid >> 5;
  const int wy   = wave >> 1;   // 0..3 : 16-row strip
  const int wx   = wave & 1;    // 0..1 : 32-col strip
  const int mn   = lane & 15;
  const int half = lane >> 4;
  const int rowBase = blockIdx.y * 64;
  const int colBase = blockIdx.x * 64;

  v8f c0 = {0.f, 0.f, 0.f, 0.f, 0.f, 0.f, 0.f, 0.f};
  v8f c1 = {0.f, 0.f, 0.f, 0.f, 0.f, 0.f, 0.f, 0.f};

  // staging coords: thread -> 16B (8 halves) of one row of each 64x32 tile
  const int sr = tid >> 2;            // 0..63
  const int sc = (tid & 3) * 8;       // 0,8,16,24
  const unsigned short* gAhi = Phi + (size_t)(rowBase + sr) * D_DIM + sc;
  const unsigned short* gAlo = Plo + (size_t)(rowBase + sr) * D_DIM + sc;
  const unsigned short* gBhi = Thi + (size_t)(colBase + sr) * D_DIM + sc;
  const unsigned short* gBlo = Tlo + (size_t)(colBase + sr) * D_DIM + sc;

  // per-thread LDS byte offsets for buffer 0 (buffer 1 = +BUF_BYTES)
  const unsigned slotOff = (unsigned)(uintptr_t)&smem[sr * LDH + sc];
  const unsigned sAhiOff = slotOff;
  const unsigned sAloOff = slotOff + 1 * ARR_BYTES;
  const unsigned sBhiOff = slotOff + 2 * ARR_BYTES;
  const unsigned sBloOff = slotOff + 3 * ARR_BYTES;

  // fragment offsets (halves, relative to smem) per ISA 16-bit A/B layout:
  // lanes 0-15 hold K={0..7,16..23}, lanes 16-31 hold K={8..15,24..31}
  const int aOff  = 0 * TILE_ELEMS + (wy * 16 + mn) * LDH + half * 8;
  const int alOff = 1 * TILE_ELEMS + (wy * 16 + mn) * LDH + half * 8;
  const int b0Off = 2 * TILE_ELEMS + (wx * 32 + mn) * LDH + half * 8;
  const int blOff = 3 * TILE_ELEMS + (wx * 32 + mn) * LDH + half * 8;
  const int b1Off = 2 * TILE_ELEMS + (wx * 32 + 16 + mn) * LDH + half * 8;
  const int b1lOff= 3 * TILE_ELEMS + (wx * 32 + 16 + mn) * LDH + half * 8;

  // prologue: DMA chunk 0 into buffer 0
  async_cp16(sAhiOff, gAhi);
  async_cp16(sAloOff, gAlo);
  async_cp16(sBhiOff, gBhi);
  async_cp16(sBloOff, gBlo);
  wait_async0();
  __syncthreads();

  for (int ch = 0; ch < NCHUNK; ++ch) {
    // prefetch chunk ch+1 into the other buffer (overlaps the WMMAs below)
    if (ch + 1 < NCHUNK) {
      const unsigned bo = ((ch + 1) & 1) * BUF_BYTES;
      const int k1 = (ch + 1) * KCB;
      async_cp16(sAhiOff + bo, gAhi + k1);
      async_cp16(sAloOff + bo, gAlo + k1);
      async_cp16(sBhiOff + bo, gBhi + k1);
      async_cp16(sBloOff + bo, gBlo + k1);
    }
    const unsigned short* base = smem + (ch & 1) * (4 * TILE_ELEMS);
    v16bf ah  = ldfrag(base + aOff);
    v16bf al  = ldfrag(base + alOff);
    v16bf b0h = ldfrag(base + b0Off);
    v16bf b0l = ldfrag(base + blOff);
    v16bf b1h = ldfrag(base + b1Off);
    v16bf b1l = ldfrag(base + b1lOff);

    c0 = __builtin_amdgcn_wmma_f32_16x16x32_bf16(false, ah, false, b0h, (short)0, c0, false, false);
    c1 = __builtin_amdgcn_wmma_f32_16x16x32_bf16(false, ah, false, b1h, (short)0, c1, false, false);
    c0 = __builtin_amdgcn_wmma_f32_16x16x32_bf16(false, ah, false, b0l, (short)0, c0, false, false);
    c1 = __builtin_amdgcn_wmma_f32_16x16x32_bf16(false, ah, false, b1l, (short)0, c1, false, false);
    c0 = __builtin_amdgcn_wmma_f32_16x16x32_bf16(false, al, false, b0h, (short)0, c0, false, false);
    c1 = __builtin_amdgcn_wmma_f32_16x16x32_bf16(false, al, false, b1h, (short)0, c1, false, false);

    wait_async0();     // next buffer resident before anyone crosses the barrier
    __syncthreads();   // includes s_wait_dscnt: all frag reads of this buf done
  }

  // ---- epilogue ----
  const int gRow0 = rowBase + wy * 16 + half * 8;
  const int gCol0 = colBase + wx * 32 + mn;
  const int gCol1 = gCol0 + 16;
  const float sb0 = sB[gCol0];
  const float sb1 = sB[gCol1];
  const float4 sa0 = *(const float4*)(sA + gRow0);
  const float4 sa1 = *(const float4*)(sA + gRow0 + 4);
  const float sav[8] = {sa0.x, sa0.y, sa0.z, sa0.w, sa1.x, sa1.y, sa1.z, sa1.w};

  float rowAcc[8];
  float colAcc0 = 0.f, colAcc1 = 0.f;
#pragma unroll
  for (int v = 0; v < 8; ++v) {
    int gr = gRow0 + v;
    float l0 = c0[v] * sav[v] * sb0;
    float l1 = c1[v] * sav[v] * sb1;
    if (gr == gCol0) diag[gr] = l0;
    if (gr == gCol1) diag[gr] = l1;
    float e0 = expf(l0), e1 = expf(l1);
    rowAcc[v] = e0 + e1;               // same row, two column tiles
    colAcc0 += e0;
    colAcc1 += e1;
  }
#pragma unroll
  for (int v = 0; v < 8; ++v) {
    float r = rowAcc[v];
    r += __shfl_xor(r, 1, 32);
    r += __shfl_xor(r, 2, 32);
    r += __shfl_xor(r, 4, 32);
    r += __shfl_xor(r, 8, 32);
    if (mn == 0) atomicAdd(&rowsum[gRow0 + v], r);
  }
  float cc0 = colAcc0 + __shfl_xor(colAcc0, 16, 32);
  float cc1 = colAcc1 + __shfl_xor(colAcc1, 16, 32);
  if (lane < 16) {
    atomicAdd(&colsum[gCol0], cc0);
    atomicAdd(&colsum[gCol1], cc1);
  }
}

// ---------------------------------------------------------------------------
// Kernel 2c (fallback, small-ws): fp32 WMMA 16x16x4 tiled GEMM.
// ---------------------------------------------------------------------------
__global__ __launch_bounds__(256) void infonce_tile_f32(
    const float* __restrict__ P, const float* __restrict__ Tm,
    const float* __restrict__ sA, const float* __restrict__ sB,
    float* __restrict__ rowsum, float* __restrict__ colsum,
    float* __restrict__ diag) {
  __shared__ float Ap[64 * LDA];
  __shared__ float Bt[32 * LDA];

  const int tid  = threadIdx.x;
  const int lane = tid & 31;
  const int wave = tid >> 5;
  const int wy   = wave >> 1;
  const int wx   = wave & 1;
  const int rowBase = blockIdx.y * 64;
  const int colBase = blockIdx.x * 32;

  v8f c = {0.f, 0.f, 0.f, 0.f, 0.f, 0.f, 0.f, 0.f};

  const int mn   = lane & 15;
  const int kofs = (lane >> 4) << 1;
  const float* aRow = &Ap[(wy * 16 + mn) * LDA + kofs];
  const float* bRow = &Bt[(wx * 16 + mn) * LDA + kofs];

  for (int k0 = 0; k0 < D_DIM; k0 += KC) {
    __syncthreads();
    {
      int r = tid >> 3, c4 = tid & 7;
      float4 v = *(const float4*)(P + (size_t)(rowBase + r) * D_DIM + k0 + c4 * 4);
      *(float4*)(&Ap[r * LDA + c4 * 4]) = v;
      int i1 = tid + 256;
      r = i1 >> 3; c4 = i1 & 7;
      v = *(const float4*)(P + (size_t)(rowBase + r) * D_DIM + k0 + c4 * 4);
      *(float4*)(&Ap[r * LDA + c4 * 4]) = v;
      r = tid >> 3; c4 = tid & 7;
      v = *(const float4*)(Tm + (size_t)(colBase + r) * D_DIM + k0 + c4 * 4);
      *(float4*)(&Bt[r * LDA + c4 * 4]) = v;
    }
    __syncthreads();
#pragma unroll
    for (int kk = 0; kk < KC; kk += 4) {
      v2f a = *(const v2f*)(aRow + kk);
      v2f b = *(const v2f*)(bRow + kk);
      c = __builtin_amdgcn_wmma_f32_16x16x4_f32(false, a, false, b, (short)0, c, false, false);
    }
  }

  const int half  = lane >> 4;
  const int gRow0 = rowBase + wy * 16 + half * 8;
  const int gCol  = colBase + wx * 16 + mn;
  const float sb  = sB[gCol];
  const float4 sa0 = *(const float4*)(sA + gRow0);
  const float4 sa1 = *(const float4*)(sA + gRow0 + 4);
  const float sav[8] = {sa0.x, sa0.y, sa0.z, sa0.w, sa1.x, sa1.y, sa1.z, sa1.w};

  float ex[8];
  float colAcc = 0.f;
#pragma unroll
  for (int v = 0; v < 8; ++v) {
    int gr = gRow0 + v;
    float l = c[v] * sav[v] * sb;
    if (gr == gCol) diag[gr] = l;
    float e = expf(l);
    ex[v] = e;
    colAcc += e;
  }
#pragma unroll
  for (int v = 0; v < 8; ++v) {
    float r = ex[v];
    r += __shfl_xor(r, 1, 32);
    r += __shfl_xor(r, 2, 32);
    r += __shfl_xor(r, 4, 32);
    r += __shfl_xor(r, 8, 32);
    if (mn == 0) atomicAdd(&rowsum[gRow0 + v], r);
  }
  float cc = colAcc + __shfl_xor(colAcc, 16, 32);
  if (lane < 16) atomicAdd(&colsum[gCol], cc);
}

// ---------------------------------------------------------------------------
// Kernel 3: loss = mean_i( 0.5*(log rowsum_i + log colsum_i) - diag_i )
// ---------------------------------------------------------------------------
__global__ __launch_bounds__(256) void infonce_finalize(
    const float* __restrict__ rowsum, const float* __restrict__ colsum,
    const float* __restrict__ diag, float* __restrict__ out) {
  __shared__ float red[256];
  float acc = 0.f;
  for (int i = threadIdx.x; i < B_DIM; i += 256)
    acc += 0.5f * (logf(rowsum[i]) + logf(colsum[i])) - diag[i];
  red[threadIdx.x] = acc;
  __syncthreads();
  for (int s = 128; s > 0; s >>= 1) {
    if (threadIdx.x < s) red[threadIdx.x] += red[threadIdx.x + s];
    __syncthreads();
  }
  if (threadIdx.x == 0) out[0] = red[0] * (1.0f / (float)B_DIM);
}

// ---------------------------------------------------------------------------
extern "C" void kernel_launch(void* const* d_in, const int* in_sizes, int n_in,
                              void* d_out, int out_size, void* d_ws, size_t ws_size,
                              hipStream_t stream) {
  const float* P  = (const float*)d_in[0];
  const float* Tm = (const float*)d_in[1];
  float* out = (float*)d_out;

  // ws layout: [sA|sB|rowsum|colsum|diag] (320KB) then bf16 split arrays (64MB)
  float* w      = (float*)d_ws;
  float* sA     = w;
  float* sB     = w + B_DIM;
  float* rowsum = w + 2 * B_DIM;
  float* colsum = w + 3 * B_DIM;
  float* diag   = w + 4 * B_DIM;

  const size_t statBytes = (size_t)5 * B_DIM * sizeof(float);        // 327680 (16B aligned)
  const size_t matHalves = (size_t)B_DIM * D_DIM;                    // 8M halves
  const size_t needed    = statBytes + 4 * matHalves * sizeof(unsigned short);

  infonce_prep<<<B_DIM / 8, 256, 0, stream>>>(P, Tm, sA, sB, rowsum, colsum);

  if (ws_size >= needed) {
    unsigned short* Phi = (unsigned short*)((char*)d_ws + statBytes);
    unsigned short* Plo = Phi + matHalves;
    unsigned short* Thi = Plo + matHalves;
    unsigned short* Tlo = Thi + matHalves;
    infonce_split<<<(B_DIM * D_DIM / 4) / 256, 256, 0, stream>>>(P, Tm, Phi, Plo, Thi, Tlo);
    dim3 grid(B_DIM / 64, B_DIM / 64);
    infonce_tile_bf16<<<grid, 256, 0, stream>>>(Phi, Plo, Thi, Tlo,
                                                sA, sB, rowsum, colsum, diag);
  } else {
    dim3 grid(B_DIM / 32, B_DIM / 64);
    infonce_tile_f32<<<grid, 256, 0, stream>>>(P, Tm, sA, sB, rowsum, colsum, diag);
  }

  infonce_finalize<<<1, 256, 0, stream>>>(rowsum, colsum, diag, out);
}
